// MoE_25890062860361
// MI455X (gfx1250) — matrix operand order repaired and checked by reading, not compiled, not run on hardware
//
#include <hip/hip_runtime.h>

// ---------------------------------------------------------------------------
// MoE (4 experts, 1x1 convs as channel GEMMs) fused for MI455X / gfx1250.
// All three GEMM stages run on v_wmma_f32_16x16x32_f16 (f16 in, f32 accum).
// Intermediates (B,E,128,H,W) never leave LDS.  Expert weights are staged
// global->LDS with CDNA5 async-to-LDS DMA (ASYNCcnt).  Each wave computes
// two 16-pixel N-tiles so every A fragment feeds two WMMAs (halves LDS BW)
// and the paired accumulators fill the WMMA->VALU hazard windows.
// ---------------------------------------------------------------------------

typedef _Float16 half_t;
typedef __attribute__((ext_vector_type(16))) _Float16 v16h;
typedef __attribute__((ext_vector_type(8)))  _Float16 v8h;
typedef __attribute__((ext_vector_type(8)))  float    v8f;

#define NE    4      // experts
#define NC    64     // emb channels
#define NH    128    // hidden channels
#define NPIX  16384  // H*W
#define TILE  256    // pixels per block (8 waves * 32 px)
#define S64   72     // padded LDS stride for K=64 rows  (72*2B = 144B, 16B-aligned)
#define S128  136    // padded LDS stride for K=128 rows (136*2B = 272B, 16B-aligned)

#define WMMA16(A, B, C) \
  __builtin_amdgcn_wmma_f32_16x16x32_f16(false, (A), false, (B), (short)0, (C), false, false)

// ---- WMMA fragment loaders (CDNA5 wave32 layouts; CDNA5 ISA §7.12.2) ----
// A (16x32 f16): lane&15 = row M; lanes<16 hold K {k0..k0+7, k0+16..k0+23},
// lanes>=16 hold K {k0+8..+15, k0+24..+31}. Two 16-byte LDS loads per lane.
__device__ __forceinline__ v16h load_a(const half_t* base, int stride, int m0,
                                       int k0, int lane) {
  const int row = m0 + (lane & 15);
  const int kb  = k0 + ((lane & 16) >> 1);          // +8 for high half-wave
  const half_t* p = base + row * stride + kb;
  v8h c0 = *(const v8h*)(p);
  v8h c1 = *(const v8h*)(p + 16);
  v16h a;
#pragma unroll
  for (int i = 0; i < 8; ++i) { a[i] = c0[i]; a[i + 8] = c1[i]; }
  return a;
}

// B (32x16 f16): lane&15 = column N; lanes<16 hold K {k0..k0+7, k0+16..k0+23},
// lanes>=16 hold K {k0+8..+15, k0+24..+31}.  Source tiles are stored N-major/K-contig.
__device__ __forceinline__ v16h load_b(const half_t* base, int stride, int k0,
                                       int n0, int lane) {
  const int col = n0 + (lane & 15);
  const int kb  = k0 + ((lane & 16) >> 1);          // +8 for high half-wave
  const half_t* p = base + col * stride + kb;
  v8h c0 = *(const v8h*)(p);
  v8h c1 = *(const v8h*)(p + 16);
  v16h b;
#pragma unroll
  for (int i = 0; i < 8; ++i) { b[i] = c0[i]; b[i + 8] = c1[i]; }
  return b;
}

// CDNA5 async global->LDS DMA, 16B per lane, GVS mode (SGPR base + u32 voffset).
// Bypasses VGPR data path; tracked by ASYNCcnt (CDNA5 ISA).
__device__ __forceinline__ void async_ld16(unsigned lds_off, unsigned src_off,
                                           const void* sbase) {
  asm volatile("global_load_async_to_lds_b128 %0, %1, %2"
               :: "v"(lds_off), "v"(src_off), "s"(sbase) : "memory");
}
__device__ __forceinline__ void wait_async0() {
  asm volatile("s_wait_asynccnt 0x0" ::: "memory");
}

// ---------------------------------------------------------------------------
// Kernel 1: global average pool.  One block per (b,c) plane of 16384 floats.
// ---------------------------------------------------------------------------
__global__ __launch_bounds__(256) void gap_kernel(const float* __restrict__ x,
                                                  float* __restrict__ gap) {
  __shared__ float red[256];
  const int plane = blockIdx.x;                     // b*64 + c
  const float* p = x + (size_t)plane * NPIX;
  float s = 0.f;
  for (int i = threadIdx.x; i < NPIX; i += 256) s += p[i];
  red[threadIdx.x] = s;
  __syncthreads();
  for (int off = 128; off > 0; off >>= 1) {
    if (threadIdx.x < off) red[threadIdx.x] += red[threadIdx.x + off];
    __syncthreads();
  }
  if (threadIdx.x == 0) gap[plane] = red[0] * (1.f / (float)NPIX);
}

// ---------------------------------------------------------------------------
// Kernel 2: gate (softmax + top-2 + renorm), aux cv^2 loss, and folding of
// conv bias + eval-mode BN into per-channel scale/shift tables.
// ---------------------------------------------------------------------------
__global__ __launch_bounds__(128) void gate_bn_kernel(
    const float* __restrict__ gap, const float* __restrict__ gw,
    const float* __restrict__ gb,
    const float* __restrict__ b1, const float* __restrict__ g1,
    const float* __restrict__ be1, const float* __restrict__ m1,
    const float* __restrict__ v1,
    const float* __restrict__ b2, const float* __restrict__ g2,
    const float* __restrict__ be2, const float* __restrict__ m2,
    const float* __restrict__ v2,
    float* __restrict__ gated, float* __restrict__ scale1,
    float* __restrict__ shift1, float* __restrict__ scale2,
    float* __restrict__ shift2, float* __restrict__ aux_out) {
  const int tid = threadIdx.x;
  // BN folding: y = (Wx + b) * inv + (beta - mean*inv)
  for (int i = tid; i < NE * NH; i += 128) {
    float inv1 = g1[i] * rsqrtf(v1[i] + 1e-5f);
    scale1[i] = inv1;
    shift1[i] = (b1[i] - m1[i]) * inv1 + be1[i];
    float inv2 = g2[i] * rsqrtf(v2[i] + 1e-5f);
    scale2[i] = inv2;
    shift2[i] = (b2[i] - m2[i]) * inv2 + be2[i];
  }
  __shared__ float sg[16 * NE];
  if (tid < 16) {
    float lg[NE];
#pragma unroll
    for (int e = 0; e < NE; ++e) {
      float d = 0.f;
      for (int c = 0; c < NC; ++c) d += gap[tid * NC + c] * gw[e * NC + c];
      lg[e] = d + gb[e];
    }
    float mx = lg[0];
#pragma unroll
    for (int e = 1; e < NE; ++e) mx = fmaxf(mx, lg[e]);
    float pr[NE], ps = 0.f;
#pragma unroll
    for (int e = 0; e < NE; ++e) { pr[e] = __expf(lg[e] - mx); ps += pr[e]; }
#pragma unroll
    for (int e = 0; e < NE; ++e) pr[e] /= ps;
    // top-2 (first index wins ties, matching lax.top_k)
    int i1 = 0;
#pragma unroll
    for (int e = 1; e < NE; ++e) if (pr[e] > pr[i1]) i1 = e;
    int i2 = (i1 == 0) ? 1 : 0;
#pragma unroll
    for (int e = 0; e < NE; ++e)
      if (e != i1 && pr[e] > pr[i2]) i2 = e;
    float den = pr[i1] + pr[i2] + 1e-8f;
#pragma unroll
    for (int e = 0; e < NE; ++e) {
      float v = (e == i1 || e == i2) ? pr[e] / den : 0.f;
      sg[tid * NE + e] = v;
      gated[tid * NE + e] = v;
    }
  }
  __syncthreads();
  if (tid == 0) {
    float imp[NE] = {0.f, 0.f, 0.f, 0.f};
    for (int b = 0; b < 16; ++b)
#pragma unroll
      for (int e = 0; e < NE; ++e) imp[e] += sg[b * NE + e];
    float mean = 0.f;
#pragma unroll
    for (int e = 0; e < NE; ++e) mean += imp[e];
    mean *= 0.25f;
    float var = 0.f;
#pragma unroll
    for (int e = 0; e < NE; ++e) {
      float d = imp[e] - mean;
      var += d * d;
    }
    var *= 0.25f;
    aux_out[0] = var / (mean * mean + 1e-10f);
  }
}

// ---------------------------------------------------------------------------
// Kernel 3: one-time f32 -> f16 conversion of all expert weights into ws.
// ---------------------------------------------------------------------------
__global__ __launch_bounds__(256) void convw_kernel(
    const float* __restrict__ W1, const float* __restrict__ W2,
    const float* __restrict__ W3, half_t* __restrict__ W1h,
    half_t* __restrict__ W2h, half_t* __restrict__ W3h) {
  const int i = blockIdx.x * 256 + threadIdx.x;     // 131072 threads
  if (i < 32768)       W1h[i]          = (half_t)W1[i];
  else if (i < 98304)  W2h[i - 32768]  = (half_t)W2[i - 32768];
  else if (i < 131072) W3h[i - 98304]  = (half_t)W3[i - 98304];
}

// ---------------------------------------------------------------------------
// Kernel 4: fused 3-stage expert MLP + gated combine.
// grid = 16 batches * 64 pixel-tiles of 256px; block = 256 threads = 8 waves;
// each wave owns two adjacent 16-pixel column tiles (N=32).
// ---------------------------------------------------------------------------
__global__ __launch_bounds__(256) void moe_main(
    const float* __restrict__ x, const half_t* __restrict__ W1h,
    const half_t* __restrict__ W2h, const half_t* __restrict__ W3h,
    const float* __restrict__ scale1, const float* __restrict__ shift1,
    const float* __restrict__ scale2, const float* __restrict__ shift2,
    const float* __restrict__ b3, const float* __restrict__ gated,
    float* __restrict__ out) {
  extern __shared__ char smem[];
  half_t* sX  = (half_t*)smem;              // [256 px][72]  x tile, K-contig
  half_t* sW1 = sX  + TILE * S64;           // [128 m][72]
  half_t* sW2 = sW1 + NH * S64;             // [128 m][136]
  half_t* sW3 = sW2 + NH * S128;            // [ 64 m][136]
  half_t* sH1 = sW3 + NC * S128;            // 8 waves * [32 px][136]
  half_t* sH2 = sH1 + 8 * 32 * S128;        // 8 waves * [32 px][136]
  float*  sBN = (float*)(sH2 + 8 * 32 * S128);  // s1,sh1,s2,sh2 (128 ea) + b3(64)

  const int tid  = threadIdx.x;
  const int lane = tid & 31;
  const int wv   = tid >> 5;
  const int lh   = (lane & 16) >> 1;        // 0 or 8: C/D row offset of half-wave
  const int nc   = lane & 15;               // C/D column within 16-wide tile
  const int b    = blockIdx.x >> 6;
  const int p0   = (blockIdx.x & 63) << 8;  // 256-pixel tile origin

  // Stage 0: load x tile (64ch x 256px), transpose to pixel-major f16.
  for (int i = tid; i < NC * TILE; i += 256) {
    const int c = i >> 8, p = i & 255;      // contiguous p per c -> coalesced
    sX[p * S64 + c] = (half_t)x[(((b << 6) + c) << 14) + p0 + p];
  }
  __syncthreads();

  half_t* sH1w = sH1 + wv * 32 * S128;
  half_t* sH2w = sH2 + wv * 32 * S128;
  const float* s1  = sBN;
  const float* sh1 = sBN + 128;
  const float* s2  = sBN + 256;
  const float* sh2 = sBN + 384;
  const float* sb3 = sBN + 512;

  // x B-fragments are invariant across experts and M-tiles: load once.
  // Two N-tiles per wave: {A: px wv*32..+15, B: px wv*32+16..+31}.
  v16h bx0A = load_b(sX, S64, 0,  wv * 32,      lane);
  v16h bx1A = load_b(sX, S64, 32, wv * 32,      lane);
  v16h bx0B = load_b(sX, S64, 0,  wv * 32 + 16, lane);
  v16h bx1B = load_b(sX, S64, 32, wv * 32 + 16, lane);

  v8f oaccA[4], oaccB[4];
#pragma unroll
  for (int t = 0; t < 4; ++t) { oaccA[t] = (v8f){}; oaccB[t] = (v8f){}; }

  for (int e = 0; e < NE; ++e) {
    if (e) __syncthreads();                 // weight tiles from prev iter consumed
    // Stage this expert's weights via async global->LDS DMA (no VGPR bounce).
    for (int ch = tid; ch < NH * NC / 8; ch += 256)
      async_ld16((unsigned)(size_t)(sW1 + (ch >> 3) * S64 + ((ch & 7) << 3)),
                 (unsigned)(e * (NH * NC) + ch * 8) * 2u, W1h);
    for (int ch = tid; ch < NH * NH / 8; ch += 256)
      async_ld16((unsigned)(size_t)(sW2 + (ch >> 4) * S128 + ((ch & 15) << 3)),
                 (unsigned)(e * (NH * NH) + ch * 8) * 2u, W2h);
    for (int ch = tid; ch < NC * NH / 8; ch += 256)
      async_ld16((unsigned)(size_t)(sW3 + (ch >> 4) * S128 + ((ch & 15) << 3)),
                 (unsigned)(e * (NC * NH) + ch * 8) * 2u, W3h);
    for (int i = tid; i < 576; i += 256) {
      float v;
      if      (i < 128) v = scale1[e * NH + i];
      else if (i < 256) v = shift1[e * NH + i - 128];
      else if (i < 384) v = scale2[e * NH + i - 256];
      else if (i < 512) v = shift2[e * NH + i - 384];
      else              v = b3[e * NC + i - 512];
      sBN[i] = v;
    }
    // Warm L2 with next expert's largest weight block (global_prefetch_b8).
    if (e + 1 < NE)
      __builtin_prefetch(W2h + (e + 1) * (NH * NH) + (tid << 6), 0, 1);
    wait_async0();                          // drain this wave's async DMA
    __syncthreads();

    const float gval = gated[(b << 2) + e];

    // ---- Stage 1: h1 = relu(bn1(W1 @ x))  M=128 K=64 N=32/wave ----
#pragma unroll
    for (int mt = 0; mt < 8; ++mt) {
      v16h a0 = load_a(sW1, S64, mt * 16, 0,  lane);
      v16h a1 = load_a(sW1, S64, mt * 16, 32, lane);
      v8f accA = (v8f){}, accB = (v8f){};
      accA = WMMA16(a0, bx0A, accA);
      accA = WMMA16(a1, bx1A, accA);
      accB = WMMA16(a0, bx0B, accB);        // reuses a0/a1; fills A-hazard slots
      accB = WMMA16(a1, bx1B, accB);
#pragma unroll
      for (int r = 0; r < 8; ++r) {
        const int m = mt * 16 + lh + r;
        float vA = fmaf(accA[r], s1[m], sh1[m]);
        float vB = fmaf(accB[r], s1[m], sh1[m]);
        sH1w[nc * S128 + m]        = (half_t)fmaxf(vA, 0.f);
        sH1w[(nc + 16) * S128 + m] = (half_t)fmaxf(vB, 0.f);
      }
    }
    // ---- Stage 2: h2 = relu(bn2(W2 @ h1))  M=128 K=128 ----
    {
      v16h bhA[4], bhB[4];
#pragma unroll
      for (int ks = 0; ks < 4; ++ks) {
        bhA[ks] = load_b(sH1w, S128, ks * 32, 0,  lane);
        bhB[ks] = load_b(sH1w, S128, ks * 32, 16, lane);
      }
#pragma unroll
      for (int mt = 0; mt < 8; ++mt) {
        v8f accA = (v8f){}, accB = (v8f){};
#pragma unroll
        for (int ks = 0; ks < 4; ++ks) {
          v16h a = load_a(sW2, S128, mt * 16, ks * 32, lane);
          accA = WMMA16(a, bhA[ks], accA);
          accB = WMMA16(a, bhB[ks], accB);
        }
#pragma unroll
        for (int r = 0; r < 8; ++r) {
          const int m = mt * 16 + lh + r;
          float vA = fmaf(accA[r], s2[m], sh2[m]);
          float vB = fmaf(accB[r], s2[m], sh2[m]);
          sH2w[nc * S128 + m]        = (half_t)fmaxf(vA, 0.f);
          sH2w[(nc + 16) * S128 + m] = (half_t)fmaxf(vB, 0.f);
        }
      }
    }
    // ---- Stage 3: out += g_e * (W3 @ h2 + b3)  M=64 K=128 ----
    {
      v16h bhA[4], bhB[4];
#pragma unroll
      for (int ks = 0; ks < 4; ++ks) {
        bhA[ks] = load_b(sH2w, S128, ks * 32, 0,  lane);
        bhB[ks] = load_b(sH2w, S128, ks * 32, 16, lane);
      }
#pragma unroll
      for (int mt = 0; mt < 4; ++mt) {
        v8f accA = (v8f){}, accB = (v8f){};
#pragma unroll
        for (int ks = 0; ks < 4; ++ks) {
          v16h a = load_a(sW3, S128, mt * 16, ks * 32, lane);
          accA = WMMA16(a, bhA[ks], accA);
          accB = WMMA16(a, bhB[ks], accB);
        }
#pragma unroll
        for (int r = 0; r < 8; ++r) {
          const float bias = sb3[mt * 16 + lh + r];
          oaccA[mt][r] += gval * (accA[r] + bias);
          oaccB[mt][r] += gval * (accB[r] + bias);
        }
      }
    }
  }

  // Write combined output: lanes 0-15 of each half-wave store 16 contiguous px.
#pragma unroll
  for (int mt = 0; mt < 4; ++mt)
#pragma unroll
    for (int r = 0; r < 8; ++r) {
      const int c = mt * 16 + lh + r;
      const int base = (((b << 6) + c) << 14) + p0 + (wv << 5) + nc;
      out[base]      = oaccA[mt][r];
      out[base + 16] = oaccB[mt][r];
    }
}

// ---------------------------------------------------------------------------
extern "C" void kernel_launch(void* const* d_in, const int* in_sizes, int n_in,
                              void* d_out, int out_size, void* d_ws,
                              size_t ws_size, hipStream_t stream) {
  const float* x   = (const float*)d_in[0];
  const float* W1  = (const float*)d_in[1];
  const float* b1  = (const float*)d_in[2];
  const float* g1  = (const float*)d_in[3];
  const float* be1 = (const float*)d_in[4];
  const float* m1  = (const float*)d_in[5];
  const float* v1  = (const float*)d_in[6];
  const float* W2  = (const float*)d_in[7];
  const float* b2  = (const float*)d_in[8];
  const float* g2  = (const float*)d_in[9];
  const float* be2 = (const float*)d_in[10];
  const float* m2  = (const float*)d_in[11];
  const float* v2  = (const float*)d_in[12];
  const float* W3  = (const float*)d_in[13];
  const float* b3  = (const float*)d_in[14];
  const float* gw  = (const float*)d_in[15];
  const float* gb  = (const float*)d_in[16];
  float* out = (float*)d_out;

  // Workspace carve-up (floats first, then 16B-aligned f16 weights).
  float* gap    = (float*)d_ws;             // 1024
  float* gated  = gap + 1024;               // 64
  float* scale1 = gated + 64;               // 512
  float* shift1 = scale1 + 512;
  float* scale2 = shift1 + 512;
  float* shift2 = scale2 + 512;             // ends @ 12544 B (16B aligned)
  half_t* W1h = (half_t*)(shift2 + 512);    // 32768 halfs
  half_t* W2h = W1h + 32768;                // 65536 halfs
  half_t* W3h = W2h + 65536;                // 32768 halfs

  gap_kernel<<<dim3(16 * NC), dim3(256), 0, stream>>>(x, gap);
  gate_bn_kernel<<<dim3(1), dim3(128), 0, stream>>>(
      gap, gw, gb, b1, g1, be1, m1, v1, b2, g2, be2, m2, v2, gated, scale1,
      shift1, scale2, shift2, out + (out_size - 1));
  convw_kernel<<<dim3(512), dim3(256), 0, stream>>>(W1, W2, W3, W1h, W2h, W3h);

  const size_t lds_bytes =
      (size_t)(TILE * S64 + NH * S64 + NH * S128 + NC * S128 +
               2 * 8 * 32 * S128) * sizeof(half_t) + 576 * sizeof(float);
  // CDNA5 WGP has 320KB LDS; raise the per-kernel dynamic-LDS cap (idempotent,
  // not a stream op, graph-capture safe).  ~249KB -> one block per WGP.
  hipFuncSetAttribute((const void*)moe_main,
                      hipFuncAttributeMaxDynamicSharedMemorySize,
                      (int)lds_bytes);
  moe_main<<<dim3(16 * 64), dim3(256), lds_bytes, stream>>>(
      x, W1h, W2h, W3h, scale1, shift1, scale2, shift2, b3, gated, out);
}
